// OptimizedFlashAttention_9131100471597
// MI455X (gfx1250) — compile-verified, matrix operand order checked by reading, not run on hardware
//
#include <hip/hip_runtime.h>
#include <math.h>

typedef __attribute__((ext_vector_type(16))) __bf16 v16bf;
typedef __attribute__((ext_vector_type(8)))  __bf16 v8bf;
typedef __attribute__((ext_vector_type(8)))  float  v8f;

#define BB 2
#define SS 2048
#define DD 1024
#define HH 16
#define DK 64
#define MROWS (BB*SS) /* 4096 */

// ---------- DPP16 cross-lane helpers (stay within a 16-lane DPP row) ----------
template<int CTRL>
static __device__ inline float dppmov(float x) {
    return __int_as_float(
        __builtin_amdgcn_update_dpp(0, __float_as_int(x), CTRL, 0xF, 0xF, true));
}
// full 16-lane butterfly: xor1, xor2, half-mirror, mirror
static __device__ inline float red_max16(float x) {
    x = fmaxf(x, dppmov<0xB1>(x));   // quad_perm(1,0,3,2)
    x = fmaxf(x, dppmov<0x4E>(x));   // quad_perm(2,3,0,1)
    x = fmaxf(x, dppmov<0x141>(x));  // ROW_HALF_MIRROR
    x = fmaxf(x, dppmov<0x140>(x));  // ROW_MIRROR
    return x;
}
static __device__ inline float red_sum16(float x) {
    x += dppmov<0xB1>(x);
    x += dppmov<0x4E>(x);
    x += dppmov<0x141>(x);
    x += dppmov<0x140>(x);
    return x;
}

// Load a 16x32 (A) or 32x16 (B) bf16 WMMA fragment: per lane, two contiguous
// 16B chunks at K offsets (k0 + 8*hf) and (k0 + 16 + 8*hf) within one row.
static __device__ inline v16bf load_frag(const __bf16* base, int k0, int hf) {
    v8bf lo = *reinterpret_cast<const v8bf*>(base + k0 + 8*hf);
    v8bf hi = *reinterpret_cast<const v8bf*>(base + k0 + 16 + 8*hf);
    return __builtin_shufflevector(lo, hi, 0,1,2,3,4,5,6,7,8,9,10,11,12,13,14,15);
}

static __device__ inline v8f wmma_bf16(v16bf a, v16bf b, v8f c) {
    return __builtin_amdgcn_wmma_f32_16x16x32_bf16(false, a, false, b, (short)0, c, false, false);
}

// One async DMA: 16 bytes/lane, global -> LDS, tracked by ASYNCcnt.
static __device__ inline void async_cp16(unsigned ldsoff, unsigned long long gaddr) {
    asm volatile("global_load_async_to_lds_b128 %0, %1, off"
                 :: "v"(ldsoff), "v"(gaddr) : "memory");
}

__global__ void f2bf_kernel(const float* __restrict__ in, __bf16* __restrict__ out, int n) {
    int i = blockIdx.x * 256 + threadIdx.x;
    if (i < n) out[i] = (__bf16)in[i];
}

// y[m][n] = sum_k A[m][k] * W[n][k].  A: bf16 [4096,1024], W: bf16 [1024,1024].
// One wave -> 16x64 output tile. MODE 0: Q (RoPE * 0.125, [B,H,S,dk] bf16)
//                                MODE 1: K (RoPE, [B,H,S,dk] bf16)
//                                MODE 2: V (transposed [B,H,dk,S] bf16)
//                                MODE 3: O (fp32 [B,S,D])
template<int MODE>
__global__ __launch_bounds__(128)
void gemm16x64(const __bf16* __restrict__ A, const __bf16* __restrict__ W,
               void* __restrict__ outp)
{
    const int lane = threadIdx.x & 31;
    const int hf   = lane >> 4;
    const int lcol = lane & 15;
    const int waveId = blockIdx.x * 4 + (threadIdx.x >> 5);
    const int mT = waveId >> 4;   // 0..255
    const int nT = waveId & 15;   // 0..15

    const __bf16* arow  = A + (size_t)(mT*16 + lcol) * DD;
    const __bf16* wrow0 = W + (size_t)(nT*64 +  0 + lcol) * DD;
    const __bf16* wrow1 = W + (size_t)(nT*64 + 16 + lcol) * DD;
    const __bf16* wrow2 = W + (size_t)(nT*64 + 32 + lcol) * DD;
    const __bf16* wrow3 = W + (size_t)(nT*64 + 48 + lcol) * DD;

    v8f acc[4] = { {}, {}, {}, {} };
    for (int k0 = 0; k0 < DD; k0 += 32) {
        __builtin_prefetch(arow + k0 + 128, 0, 1);
        v16bf af = load_frag(arow, k0, hf);
        acc[0] = wmma_bf16(af, load_frag(wrow0, k0, hf), acc[0]);
        acc[1] = wmma_bf16(af, load_frag(wrow1, k0, hf), acc[1]);
        acc[2] = wmma_bf16(af, load_frag(wrow2, k0, hf), acc[2]);
        acc[3] = wmma_bf16(af, load_frag(wrow3, k0, hf), acc[3]);
    }

    if (MODE == 3) {
        float* out = (float*)outp;
        #pragma unroll
        for (int j = 0; j < 4; ++j) {
            const int n = nT*64 + 16*j + lcol;
            #pragma unroll
            for (int r = 0; r < 8; ++r) {
                const int m = mT*16 + r + 8*hf;
                out[(size_t)m * DD + n] = acc[j][r];
            }
        }
    } else if (MODE == 2) {
        __bf16* out = (__bf16*)outp;
        const int h = nT;
        #pragma unroll
        for (int j = 0; j < 4; ++j) {
            const int d = 16*j + lcol;
            #pragma unroll
            for (int r = 0; r < 8; ++r) {
                const int m = mT*16 + r + 8*hf;
                const int b = m >> 11;
                const int s = m & 2047;
                out[((size_t)(b*HH + h)*DK + d) * SS + s] = (__bf16)acc[j][r];
            }
        }
    } else {  // Q or K with RoPE
        __bf16* out = (__bf16*)outp;
        const int h = nT;                         // 64-wide tile == one head
        const int b = (mT*16) >> 11;
        const int baseS = (mT*16 + 8*hf) & 2047;  // s for r=0
        #pragma unroll
        for (int j = 0; j < 4; ++j) {
            const int d = 16*j + lcol;            // 0..63 within head
            const float fr = __powf(10000.0f, -(float)(d & ~1) * (1.0f/64.0f));
            // sincos recurrence: s advances by 1 per row r
            float sn = sinf((float)baseS * fr), cs = cosf((float)baseS * fr);
            const float sf = sinf(fr), cf = cosf(fr);
            #pragma unroll
            for (int r = 0; r < 8; ++r) {
                const int s = baseS + r;
                const float v  = acc[j][r];
                const float pv = dppmov<0xB1>(v);   // RoPE partner (d^1), adjacent lane
                float res = (lcol & 1) ? (v*cs + pv*sn) : (v*cs - pv*sn);
                if (MODE == 0) res *= 0.125f;       // fold 1/sqrt(dk) into Q
                out[((size_t)(b*HH + h)*SS + s) * DK + d] = (__bf16)res;
                const float sn2 = sn*cf + cs*sf;    // advance angle by fr
                cs = cs*cf - sn*sf;
                sn = sn2;
            }
        }
    }
}

// Flash attention with online softmax + ALiBi.
// One wave per (b,h,16-query tile); the 4 waves of a block share (b,h) and
// stage the common K/V stream through double-buffered LDS with the async
// global->LDS engine (ASYNCcnt), overlapping DMA with WMMA compute.
__global__ __launch_bounds__(128)
void attn_kernel(const __bf16* __restrict__ Q, const __bf16* __restrict__ K,
                 const __bf16* __restrict__ VT, __bf16* __restrict__ AO)
{
    __shared__ __align__(16) __bf16 kst[2][32*64];  // K chunk: 32 keys x 64 dk (4KB each)
    __shared__ __align__(16) __bf16 vst[2][64*32];  // V chunk: 64 dk x 32 keys (4KB each)
    __shared__ __align__(16) __bf16 psm[4][16*32];  // per-wave P tile

    const int t    = threadIdx.x;          // 0..127
    const int lane = t & 31;
    const int hf   = lane >> 4;
    const int lcol = lane & 15;
    const int widx = t >> 5;
    const int waveId = blockIdx.x * 4 + widx;
    const int qT = waveId & 127;           // blocks never straddle a head (4 | 128)
    const int h  = (waveId >> 7) & 15;
    const int b  = waveId >> 11;
    const int q0 = qT * 16;

    __bf16* sm = &psm[widx][0];
    const unsigned kof[2] = { (unsigned)(size_t)&kst[0][0], (unsigned)(size_t)&kst[1][0] };
    const unsigned vof[2] = { (unsigned)(size_t)&vst[0][0], (unsigned)(size_t)&vst[1][0] };

    const __bf16* qbase = Q + ((size_t)(b*HH + h)*SS + q0 + lcol) * DK;
    const v16bf aq0 = load_frag(qbase,  0, hf);
    const v16bf aq1 = load_frag(qbase, 32, hf);

    const unsigned long long kg = (unsigned long long)(size_t)(K  + (size_t)(b*HH + h)*SS*DK);
    const unsigned long long vg = (unsigned long long)(size_t)(VT + (size_t)(b*HH + h)*DK*SS);
    const int seg = t >> 2;                // 0..31: V row handled by this thread
    const int sof = (t & 3) * 16;          // byte offset within the 64B V row chunk

    const float alibi = exp2f(-0.5f * (float)h);   // true ALiBi slope
    float rmax[8], rsum[8];
    #pragma unroll
    for (int r = 0; r < 8; ++r) { rmax[r] = -1e30f; rsum[r] = 0.0f; }
    v8f acc[4] = { {}, {}, {}, {} };

    // stage chunk kk (32 keys) into buffer buf: 4 async B128 issues per wave
    auto stage = [&](int buf, int kk) {
        const unsigned long long ksrc = kg + (unsigned long long)kk * 128ull;    // rows contiguous
        async_cp16(kof[buf] + t*16,        ksrc + t*16);
        async_cp16(kof[buf] + 2048 + t*16, ksrc + 2048 + t*16);
        const unsigned long long vsrc = vg + (unsigned long long)kk * 2ull + sof;
        async_cp16(vof[buf] + seg*64 + sof,        vsrc + (unsigned long long)seg * 4096ull);
        async_cp16(vof[buf] + 2048 + seg*64 + sof, vsrc + (unsigned long long)(seg + 32) * 4096ull);
    };

    stage(0, 0);

    for (int it = 0; it < SS/32; ++it) {
        const int kk  = it * 32;
        const int cur = it & 1;
        __syncthreads();                       // WAR: everyone done reading buf cur^1
        if (it + 1 < SS/32) {
            stage(cur ^ 1, kk + 32);
            asm volatile("s_wait_asynccnt 0x4" ::: "memory");  // chunk `it` landed (in-order)
        } else {
            asm volatile("s_wait_asynccnt 0x0" ::: "memory");
        }
        __syncthreads();                       // RAW: all waves' quarters landed

        const __bf16* kbuf = &kst[cur][0];
        const __bf16* vbuf = &vst[cur][0];
        const __bf16* k0row = kbuf + lcol * 64;
        const __bf16* k1row = kbuf + (16 + lcol) * 64;

        v8f s0 = {}, s1 = {};
        s0 = wmma_bf16(aq0, load_frag(k0row,  0, hf), s0);
        s0 = wmma_bf16(aq1, load_frag(k0row, 32, hf), s0);
        s1 = wmma_bf16(aq0, load_frag(k1row,  0, hf), s1);
        s1 = wmma_bf16(aq1, load_frag(k1row, 32, hf), s1);

        #pragma unroll
        for (int r = 0; r < 8; ++r) {
            const float qp = (float)(q0 + r + 8*hf);
            // scores already carry 1/sqrt(dk) via prescaled Q
            const float a0 = s0[r] - alibi * fabsf(qp - (float)(kk + lcol));
            const float a1 = s1[r] - alibi * fabsf(qp - (float)(kk + 16 + lcol));
            const float tm = red_max16(fmaxf(a0, a1));        // DPP butterfly
            const float nm   = fmaxf(rmax[r], tm);
            const float corr = __expf(rmax[r] - nm);
            const float p0   = __expf(a0 - nm);
            const float p1   = __expf(a1 - nm);
            const float ps   = red_sum16(p0 + p1);            // DPP butterfly
            rsum[r] = rsum[r]*corr + ps;
            rmax[r] = nm;
            acc[0][r] *= corr; acc[1][r] *= corr; acc[2][r] *= corr; acc[3][r] *= corr;
            const int m = r + 8*hf;
            sm[m*32 + lcol]      = (__bf16)p0;
            sm[m*32 + 16 + lcol] = (__bf16)p1;
        }
        __builtin_amdgcn_wave_barrier();
        asm volatile("s_wait_dscnt 0x0" ::: "memory");  // in-wave LDS RAW across lanes

        const v16bf pa = load_frag(sm + lcol*32, 0, hf);     // P as A-fragment
        acc[0] = wmma_bf16(pa, load_frag(vbuf + ( 0 + lcol)*32, 0, hf), acc[0]);
        acc[1] = wmma_bf16(pa, load_frag(vbuf + (16 + lcol)*32, 0, hf), acc[1]);
        acc[2] = wmma_bf16(pa, load_frag(vbuf + (32 + lcol)*32, 0, hf), acc[2]);
        acc[3] = wmma_bf16(pa, load_frag(vbuf + (48 + lcol)*32, 0, hf), acc[3]);
    }

    // normalize and store to [B,S,D] bf16
    #pragma unroll
    for (int r = 0; r < 8; ++r) {
        const float inv = 1.0f / rsum[r];
        const int m = r + 8*hf;
        const size_t rowoff = ((size_t)b*SS + q0 + m) * DD + (size_t)h*DK;
        AO[rowoff +  0 + lcol] = (__bf16)(acc[0][r] * inv);
        AO[rowoff + 16 + lcol] = (__bf16)(acc[1][r] * inv);
        AO[rowoff + 32 + lcol] = (__bf16)(acc[2][r] * inv);
        AO[rowoff + 48 + lcol] = (__bf16)(acc[3][r] * inv);
    }
}

extern "C" void kernel_launch(void* const* d_in, const int* in_sizes, int n_in,
                              void* d_out, int out_size, void* d_ws, size_t ws_size,
                              hipStream_t stream)
{
    (void)in_sizes; (void)n_in; (void)out_size; (void)ws_size;
    const float* x  = (const float*)d_in[0];
    const float* Wq = (const float*)d_in[1];
    const float* Wk = (const float*)d_in[2];
    const float* Wv = (const float*)d_in[3];
    const float* Wo = (const float*)d_in[4];

    char* ws = (char*)d_ws;
    size_t off = 0;
    __bf16* xb  = (__bf16*)(ws + off); off += (size_t)MROWS*DD*2;
    __bf16* wqb = (__bf16*)(ws + off); off += (size_t)DD*DD*2;
    __bf16* wkb = (__bf16*)(ws + off); off += (size_t)DD*DD*2;
    __bf16* wvb = (__bf16*)(ws + off); off += (size_t)DD*DD*2;
    __bf16* wob = (__bf16*)(ws + off); off += (size_t)DD*DD*2;
    __bf16* qb  = (__bf16*)(ws + off); off += (size_t)MROWS*DD*2;  // [B,H,S,dk], pre-scaled 0.125
    __bf16* kb  = (__bf16*)(ws + off); off += (size_t)MROWS*DD*2;  // [B,H,S,dk]
    __bf16* vtb = (__bf16*)(ws + off); off += (size_t)MROWS*DD*2;  // [B,H,dk,S]
    __bf16* aob = (__bf16*)(ws + off); off += (size_t)MROWS*DD*2;  // [B,S,D]

    const int nx = MROWS*DD, nw = DD*DD;
    f2bf_kernel<<<(nx+255)/256, 256, 0, stream>>>(x,  xb,  nx);
    f2bf_kernel<<<(nw+255)/256, 256, 0, stream>>>(Wq, wqb, nw);
    f2bf_kernel<<<(nw+255)/256, 256, 0, stream>>>(Wk, wkb, nw);
    f2bf_kernel<<<(nw+255)/256, 256, 0, stream>>>(Wv, wvb, nw);
    f2bf_kernel<<<(nw+255)/256, 256, 0, stream>>>(Wo, wob, nw);

    gemm16x64<0><<<1024, 128, 0, stream>>>(xb, wqb, (void*)qb);
    gemm16x64<1><<<1024, 128, 0, stream>>>(xb, wkb, (void*)kb);
    gemm16x64<2><<<1024, 128, 0, stream>>>(xb, wvb, (void*)vtb);

    attn_kernel<<<1024, 128, 0, stream>>>(qb, kb, vtb, aob);

    gemm16x64<3><<<1024, 128, 0, stream>>>(aob, wob, d_out);
}
